// MPNNLayer_36309653520739
// MI455X (gfx1250) — compile-verified, matrix operand order checked by reading, not run on hardware
//
#include <hip/hip_runtime.h>

typedef __bf16 bf16_t;
typedef __attribute__((ext_vector_type(16))) __bf16 v16bf;
typedef __attribute__((ext_vector_type(8)))  __bf16 v8bf;
typedef __attribute__((ext_vector_type(8)))  float   v8f;

// ---------------------------------------------------------------------------
// Fragment loader for 16-bit WMMA operands from a row-major [*, K] matrix.
// ISA 16-bit A-matrix 16x32 layout (B^T has identical layout):
//   lane 0-15 : row M=lane,    holds K = k0+[0..7]  and k0+16+[0..7]
//   lane 16-31: row M=lane-16, holds K = k0+8+[0..7] and k0+24+[0..7]
// row must be 16-byte aligned at (kbase + half*8).
// ---------------------------------------------------------------------------
__device__ __forceinline__ v16bf load_frag(const bf16_t* __restrict__ row,
                                           int kbase, int half) {
  v8bf lo = *reinterpret_cast<const v8bf*>(row + kbase + half * 8);
  v8bf hi = *reinterpret_cast<const v8bf*>(row + kbase + 16 + half * 8);
  v16bf f;
#pragma unroll
  for (int i = 0; i < 8; ++i) { f[i] = lo[i]; f[i + 8] = hi[i]; }
  return f;
}

__device__ __forceinline__ v8f wmma_bf16(v16bf a, v16bf b, v8f c) {
  return __builtin_amdgcn_wmma_f32_16x16x32_bf16(false, a, false, b, (short)0,
                                                 c, false, false);
}

// ---------------------------------------------------------------------------
// Generic wave-tile GEMM:  C[M,N] = A[M,K] @ W[N,K]^T  (bf16 in, f32 acc)
// Each wave computes a 16x64 tile (4 WMMAs per 32-wide k-step, A reused).
// K loop is software-pipelined with ping-pong fragment sets X/Y (unroll by 2)
// so next-step loads overlap current-step WMMAs with NO register rotation
// copies: X and Y swap roles each step.
// OUT_MODE: 0 = f32 out, 1 = bf16 out, 2 = bf16 transposed out (per-P-row
// batch: element (m,n) stored at [(m/P)*N + n]*P + m%P).
// Optional bias[n] and optional f32 residual addsrc[m*N+n].
// blockDim = 128 (4 waves). blockIdx.y = batch (strides in elements).
// ---------------------------------------------------------------------------
template <int OUT_MODE, bool BIAS, bool ADDSRC>
__global__ __launch_bounds__(128) void gemm_wmma(
    const bf16_t* __restrict__ A, int lda, long long strideA,
    const bf16_t* __restrict__ W, int ldw, long long strideW,
    const float* __restrict__ bias,
    const float* __restrict__ addsrc,
    void* __restrict__ Cout, long long strideC,
    int M, int N, int K, int P) {
  const int lane = threadIdx.x & 31;
  const int wave = threadIdx.x >> 5;
  const int batch = blockIdx.y;
  A += (long long)batch * strideA;
  W += (long long)batch * strideW;

  const int nt = N >> 6;  // number of 64-wide N supertiles
  const int wid = blockIdx.x * 4 + wave;
  const int tm = (wid / nt) << 4;
  const int tn = (wid % nt) << 6;
  if (tm >= M) return;

  const int half = lane >> 4;
  const int l15 = lane & 15;

  const bf16_t* arow = A + (size_t)(tm + l15) * lda;
  const bf16_t* wrow[4];
#pragma unroll
  for (int j = 0; j < 4; ++j) wrow[j] = W + (size_t)(tn + 16 * j + l15) * ldw;

  v8f acc[4] = {v8f{}, v8f{}, v8f{}, v8f{}};

  // --- ping-pong pipelined K loop (no rotation copies) ---
  v16bf aX = load_frag(arow, 0, half);
  v16bf bX[4];
#pragma unroll
  for (int j = 0; j < 4; ++j) bX[j] = load_frag(wrow[j], 0, half);

  int k = 32;
  while (k + 32 < K) {
    // prefetch Y(k) under X's WMMAs
    v16bf aY = load_frag(arow, k, half);
    v16bf bY[4];
#pragma unroll
    for (int j = 0; j < 4; ++j) bY[j] = load_frag(wrow[j], k, half);
#pragma unroll
    for (int j = 0; j < 4; ++j) acc[j] = wmma_bf16(aX, bX[j], acc[j]);
    // prefetch X(k+32) under Y's WMMAs
    aX = load_frag(arow, k + 32, half);
#pragma unroll
    for (int j = 0; j < 4; ++j) bX[j] = load_frag(wrow[j], k + 32, half);
#pragma unroll
    for (int j = 0; j < 4; ++j) acc[j] = wmma_bf16(aY, bY[j], acc[j]);
    k += 64;
  }
  if (k < K) {  // one trailing step beyond X
    v16bf aY = load_frag(arow, k, half);
    v16bf bY[4];
#pragma unroll
    for (int j = 0; j < 4; ++j) bY[j] = load_frag(wrow[j], k, half);
#pragma unroll
    for (int j = 0; j < 4; ++j) acc[j] = wmma_bf16(aX, bX[j], acc[j]);
#pragma unroll
    for (int j = 0; j < 4; ++j) acc[j] = wmma_bf16(aY, bY[j], acc[j]);
  } else {
#pragma unroll
    for (int j = 0; j < 4; ++j) acc[j] = wmma_bf16(aX, bX[j], acc[j]);
  }

  // Epilogue. C/D layout: VGPR r, lane L -> m = tm + r + 8*(L>=16), n = tn+(L&15).
#pragma unroll
  for (int j = 0; j < 4; ++j) {
    const int n = tn + 16 * j + l15;
    const float bn = BIAS ? bias[n] : 0.0f;
    if constexpr (OUT_MODE == 2) {
      const int bb = tm / P;
      const int v0 = (tm % P) + 8 * half;
      bf16_t* dst = (bf16_t*)Cout + ((size_t)bb * N + n) * P + v0;
      v8bf t;
#pragma unroll
      for (int r = 0; r < 8; ++r) t[r] = (bf16_t)(acc[j][r] + bn);
      *reinterpret_cast<v8bf*>(dst) = t;
    } else {
#pragma unroll
      for (int r = 0; r < 8; ++r) {
        const int m = tm + r + 8 * half;
        float val = acc[j][r] + bn;
        if constexpr (ADDSRC) val += addsrc[(size_t)m * N + n];
        const size_t idx = (size_t)((long long)batch * strideC) + (size_t)m * N + n;
        if constexpr (OUT_MODE == 0) ((float*)Cout)[idx] = val;
        else ((bf16_t*)Cout)[idx] = (bf16_t)val;
      }
    }
  }
}

// ---------------------------------------------------------------------------
// Elementwise / prep kernels
// ---------------------------------------------------------------------------
__global__ void f2bf_kernel(const float* __restrict__ src, bf16_t* __restrict__ dst,
                            long long n) {
  long long i = (long long)blockIdx.x * blockDim.x + threadIdx.x;
  if (i < n) dst[i] = (bf16_t)src[i];
}

// per (b,v): degree + adjacency row to bf16
__global__ void deg_adj_kernel(const int* __restrict__ adj, float* __restrict__ deg,
                               bf16_t* __restrict__ adj_bf, long long M, int V) {
  long long i = (long long)blockIdx.x * blockDim.x + threadIdx.x;
  if (i >= M) return;
  const int* ap = adj + i * V;
  bf16_t* op = adj_bf + i * V;
  float s = 0.f;
  for (int w = 0; w < V; ++w) {
    float a = (float)ap[w];
    s += a;
    op[w] = (bf16_t)a;
  }
  deg[i] = s;
}

// agg_e[b,v,e] = sum_w adj[b,v,w] * edges[b,v,w,e]  (head-independent)
__global__ void agg_edges_kernel(const float* __restrict__ edges,
                                 const int* __restrict__ adj,
                                 bf16_t* __restrict__ agg_bf, long long M, int V) {
  long long i = (long long)blockIdx.x * blockDim.x + threadIdx.x;
  if (i >= M * 32) return;
  const int e = (int)(i & 31);
  const long long bv = i >> 5;
  const float* ep = edges + bv * (long long)V * 32 + e;
  const int* ap = adj + bv * V;
  float s = 0.f;
  for (int w = 0; w < V; ++w)
    if (ap[w]) s += ep[(long long)w * 32];
  agg_bf[i] = (bf16_t)s;
}

// msg = relu(tv*deg + nbr + ewc + b2*deg) -> bf16
__global__ void msg_kernel(const float* __restrict__ tv, const float* __restrict__ nbr,
                           const float* __restrict__ ewc, const float* __restrict__ deg,
                           const float* __restrict__ b2, bf16_t* __restrict__ msg_bf,
                           long long M) {
  long long i = (long long)blockIdx.x * blockDim.x + threadIdx.x;
  if (i >= M * 256) return;
  const long long m = i >> 8;
  const int n = (int)(i & 255);
  const float d = deg[m];
  float x = tv[i] * d + nbr[i] + ewc[i] + b2[n] * d;
  msg_bf[i] = (bf16_t)(x > 0.f ? x : 0.f);
}

__device__ __forceinline__ float sigmoidf(float x) { return 1.f / (1.f + expf(-x)); }

// GRU gates (torch order r,z,n) + head relu, write into upd[:, h*256+n]
__global__ void gru_kernel(const float* __restrict__ gi, const float* __restrict__ gh,
                           const float* __restrict__ nodes, bf16_t* __restrict__ upd,
                           int hofs, long long M) {
  long long i = (long long)blockIdx.x * blockDim.x + threadIdx.x;
  if (i >= M * 256) return;
  const long long m = i >> 8;
  const int n = (int)(i & 255);
  const float* gim = gi + m * 768;
  const float* ghm = gh + m * 768;
  const float r = sigmoidf(gim[n] + ghm[n]);
  const float z = sigmoidf(gim[256 + n] + ghm[256 + n]);
  const float nn = tanhf(gim[512 + n] + r * ghm[512 + n]);
  const float h = (1.f - z) * nn + z * nodes[i];
  upd[m * 1024 + hofs + n] = (bf16_t)(h > 0.f ? h : 0.f);
}

static inline dim3 ewg(long long n) { return dim3((unsigned)((n + 255) / 256)); }

// ---------------------------------------------------------------------------
extern "C" void kernel_launch(void* const* d_in, const int* in_sizes, int n_in,
                              void* d_out, int out_size, void* d_ws, size_t ws_size,
                              hipStream_t stream) {
  (void)in_sizes; (void)n_in; (void)out_size; (void)ws_size;
  constexpr int B = 64, V = 96, NH = 4;
  constexpr long long M = (long long)B * V;  // 6144

  const float* nodes = (const float*)d_in[0];
  const float* edges = (const float*)d_in[1];
  const int*   adj   = (const int*)d_in[2];
  const float* W1    = (const float*)d_in[3];
  const float* b1    = (const float*)d_in[4];
  const float* W2    = (const float*)d_in[5];
  const float* b2    = (const float*)d_in[6];
  const float* Wih   = (const float*)d_in[7];
  const float* Whh   = (const float*)d_in[8];
  const float* bih   = (const float*)d_in[9];
  const float* bhh   = (const float*)d_in[10];
  const float* Wout  = (const float*)d_in[11];
  const float* bout  = (const float*)d_in[12];
  float* out = (float*)d_out;

  // --- workspace carve-up (256B aligned) ---
  char* base = (char*)d_ws;
  size_t off = 0;
  auto alloc = [&](size_t bytes) -> char* {
    char* p = base + off;
    off += (bytes + 255) & ~(size_t)255;
    return p;
  };
  bf16_t* nodes_bf = (bf16_t*)alloc(M * 256 * 2);
  bf16_t* W1_bf    = (bf16_t*)alloc((size_t)NH * 256 * 256 * 2);
  bf16_t* W2_bf    = (bf16_t*)alloc((size_t)NH * 256 * 544 * 2);
  bf16_t* Wih_bf   = (bf16_t*)alloc((size_t)NH * 768 * 256 * 2);
  bf16_t* Whh_bf   = (bf16_t*)alloc((size_t)NH * 768 * 256 * 2);
  bf16_t* Wout_bf  = (bf16_t*)alloc((size_t)256 * 1024 * 2);
  bf16_t* adj_bf   = (bf16_t*)alloc(M * V * 2);
  float*  deg      = (float*)alloc(M * 4);
  bf16_t* agg_bf   = (bf16_t*)alloc(M * 32 * 2);
  bf16_t* hv_bf    = (bf16_t*)alloc(M * 256 * 2);
  bf16_t* twT      = (bf16_t*)alloc((size_t)B * 256 * V * 2);
  bf16_t* msg_bf   = (bf16_t*)alloc(M * 256 * 2);
  bf16_t* upd_bf   = (bf16_t*)alloc(M * 1024 * 2);
  float*  regX     = (float*)alloc(M * 768 * 4);  // tv|nbr|ewc, later reused as gi
  float*  gh       = (float*)alloc(M * 768 * 4);
  float* tv  = regX;
  float* nbr = regX + M * 256;
  float* ewc = regX + 2 * M * 256;
  float* gi  = regX;

  // --- prep: convert inputs/weights to bf16, degree, edge aggregation ---
  f2bf_kernel<<<ewg(M * 256), 256, 0, stream>>>(nodes, nodes_bf, M * 256);
  f2bf_kernel<<<ewg(NH * 256 * 256), 256, 0, stream>>>(W1, W1_bf, (long long)NH * 256 * 256);
  f2bf_kernel<<<ewg(NH * 256 * 544), 256, 0, stream>>>(W2, W2_bf, (long long)NH * 256 * 544);
  f2bf_kernel<<<ewg(NH * 768 * 256), 256, 0, stream>>>(Wih, Wih_bf, (long long)NH * 768 * 256);
  f2bf_kernel<<<ewg(NH * 768 * 256), 256, 0, stream>>>(Whh, Whh_bf, (long long)NH * 768 * 256);
  f2bf_kernel<<<ewg(256 * 1024), 256, 0, stream>>>(Wout, Wout_bf, 256 * 1024);
  deg_adj_kernel<<<ewg(M), 256, 0, stream>>>(adj, deg, adj_bf, M, V);
  agg_edges_kernel<<<ewg(M * 32), 256, 0, stream>>>(edges, adj, agg_bf, M, V);

  const dim3 g256(384);    // M=6144, N=256 -> 1536 wave-tiles / 4
  const dim3 g768(1152);   // M=6144, N=768 -> 4608 wave-tiles / 4
  const dim3 gnbr(6, 64);  // per batch: M=96,N=256 -> 24 wave-tiles

  for (int h = 0; h < NH; ++h) {
    const bf16_t* W1h  = W1_bf + (size_t)h * 256 * 256;
    const bf16_t* Wa   = W2_bf + (size_t)h * 256 * 544;        // cols [0,256)
    const bf16_t* Wb   = Wa + 256;                             // cols [256,512)
    const bf16_t* Wc   = Wa + 512;                             // cols [512,544)
    const bf16_t* Wihh = Wih_bf + (size_t)h * 768 * 256;
    const bf16_t* Whhh = Whh_bf + (size_t)h * 768 * 256;

    // hv = bf16(nodes @ W1h^T + b1h)
    gemm_wmma<1, true, false><<<g256, 128, 0, stream>>>(
        nodes_bf, 256, 0, W1h, 256, 0, b1 + h * 256, nullptr, hv_bf, 0,
        (int)M, 256, 256, 1);
    // tv = hv @ Wa^T (f32)
    gemm_wmma<0, false, false><<<g256, 128, 0, stream>>>(
        hv_bf, 256, 0, Wa, 544, 0, nullptr, nullptr, tv, 0, (int)M, 256, 256, 1);
    // twT[b, m, w] = bf16(hv @ Wb^T) stored transposed per batch
    gemm_wmma<2, false, false><<<g256, 128, 0, stream>>>(
        hv_bf, 256, 0, Wb, 544, 0, nullptr, nullptr, twT, 0, (int)M, 256, 256, V);
    // nbr[b] = adj[b] @ tw[b]  ==  adj[b] @ twT[b]^T  (batched)
    gemm_wmma<0, false, false><<<gnbr, 128, 0, stream>>>(
        adj_bf, V, (long long)V * V, twT, V, (long long)256 * V, nullptr, nullptr,
        nbr, (long long)V * 256, V, 256, V, 1);
    // ewc = agg_e @ Wc^T  (K=32)
    gemm_wmma<0, false, false><<<g256, 128, 0, stream>>>(
        agg_bf, 32, 0, Wc, 544, 0, nullptr, nullptr, ewc, 0, (int)M, 256, 32, 1);
    // msg = relu(tv*deg + nbr + ewc + b2*deg)
    msg_kernel<<<ewg(M * 256), 256, 0, stream>>>(tv, nbr, ewc, deg, b2 + h * 256,
                                                 msg_bf, M);
    // gi = msg @ Wih^T + bih   (overwrites tv/nbr/ewc region)
    gemm_wmma<0, true, false><<<g768, 128, 0, stream>>>(
        msg_bf, 256, 0, Wihh, 256, 0, bih + h * 768, nullptr, gi, 0,
        (int)M, 768, 256, 1);
    // gh = nodes @ Whh^T + bhh
    gemm_wmma<0, true, false><<<g768, 128, 0, stream>>>(
        nodes_bf, 256, 0, Whhh, 256, 0, bhh + h * 768, nullptr, gh, 0,
        (int)M, 768, 256, 1);
    // GRU + relu -> upd column block h
    gru_kernel<<<ewg(M * 256), 256, 0, stream>>>(gi, gh, nodes, upd_bf, h * 256, M);
  }

  // out = nodes + relu(upd) @ Wout^T + bout
  gemm_wmma<0, true, true><<<g256, 128, 0, stream>>>(
      upd_bf, 1024, 0, Wout_bf, 1024, 0, bout, nodes, out, 0, (int)M, 256, 1024, 1);
}
